// ProductKeyMemory_7172595384952
// MI455X (gfx1250) — compile-verified, matrix operand order checked by reading, not run on hardware
//
#include <hip/hip_runtime.h>

typedef __bf16 bf16_t;
typedef __attribute__((ext_vector_type(16))) __bf16 v16bf;
typedef __attribute__((ext_vector_type(8)))  float  v8f;

#define DIM      1024
#define HALF     512
#define NUM_KEYS 512
#define TOPK     32
#define NTOK     4096   // B*S = 2*2048

// ---------------------------------------------------------------------------
// fp32 -> bf16 conversion (one-shot per launch; weights + activations)
// ---------------------------------------------------------------------------
__global__ __launch_bounds__(256) void cvt_f32_bf16(const float* __restrict__ in,
                                                    bf16_t* __restrict__ out, int n) {
  int i = blockIdx.x * 256 + threadIdx.x;
  if (i < n) out[i] = (bf16_t)in[i];
}

// ---------------------------------------------------------------------------
// bf16 WMMA GEMM:  C[M x N] = A[M x K] * B[N x K]^T (+bias) with epilogue modes
//   mode 0: fp32 store, +bias
//   mode 1: fp32 store, +bias, SiLU
//   mode 2: fp32 store, no bias
//   mode 3: bf16 store, +bias
// Block: 256 threads = 8 waves; each wave computes a 32x64 strip (2 A-frags x
// 4 B-frags = 8 WMMAs per k-step); block covers 256x64.
// grid.x = N/64, grid.y = M/256.
// ---------------------------------------------------------------------------
__global__ __launch_bounds__(256) void gemm_bf16_wmma(
    const bf16_t* __restrict__ A, int lda,
    const bf16_t* __restrict__ B, int ldb,
    const float*  __restrict__ bias,
    void* __restrict__ Cv, int ldc,
    int K, int mode) {
  const int lane  = threadIdx.x & 31;
  const int wid   = threadIdx.x >> 5;
  const int m0    = blockIdx.y * 256 + wid * 32;
  const int n0    = blockIdx.x * 64;
  const int mrow  = lane & 15;   // row within tile (A) / col within tile (B)
  const int khalf = lane >> 4;   // which 16-element K half this lane holds

  v8f acc[2][4] = {};

  const bf16_t* aptr0 = A + (size_t)(m0 + mrow) * lda + khalf * 16;
  const bf16_t* aptr1 = aptr0 + (size_t)16 * lda;
  const bf16_t* bptr  = B + (size_t)(n0 + mrow) * ldb + khalf * 16;

  for (int k = 0; k < K; k += 32) {
    v16bf a0 = *(const v16bf*)(aptr0 + k);
    v16bf a1 = *(const v16bf*)(aptr1 + k);
#pragma unroll
    for (int nn = 0; nn < 4; ++nn) {
      v16bf bfv = *(const v16bf*)(bptr + (size_t)nn * 16 * ldb + k);
      acc[0][nn] = __builtin_amdgcn_wmma_f32_16x16x32_bf16(
          false, a0, false, bfv, (short)0, acc[0][nn], false, false);
      acc[1][nn] = __builtin_amdgcn_wmma_f32_16x16x32_bf16(
          false, a1, false, bfv, (short)0, acc[1][nn], false, false);
    }
  }

  // C/D layout: VGPR v, lane l -> row = v + 8*(l>>4), col = l&15
#pragma unroll
  for (int mm = 0; mm < 2; ++mm) {
    const int rowBase = m0 + mm * 16 + 8 * khalf;
#pragma unroll
    for (int nn = 0; nn < 4; ++nn) {
      const int col = n0 + nn * 16 + mrow;
      const float b = bias ? bias[col] : 0.0f;
#pragma unroll
      for (int v = 0; v < 8; ++v) {
        const int row = rowBase + v;
        float val = acc[mm][nn][v] + b;
        if (mode == 1) val = val / (1.0f + __expf(-val));   // SiLU
        if (mode == 3) ((bf16_t*)Cv)[(size_t)row * ldc + col] = (bf16_t)val;
        else           ((float*)Cv)[(size_t)row * ldc + col] = val;
      }
    }
  }
}

// ---------------------------------------------------------------------------
// Per-token: top-32 of scores1 (wave0) and scores2 (wave1) in parallel,
// 32x32 combine top-32 (wave0), softmax, then block-wide coalesced gather of
// the 32 value rows, multiply by SiLU gate, store bf16 for the final GEMM.
// One 256-thread block per token.
// ---------------------------------------------------------------------------
__global__ __launch_bounds__(256) void pkm_topk_gather(
    const float* __restrict__ s1f, const float* __restrict__ s2f,
    const float* __restrict__ values, const float* __restrict__ gatef,
    bf16_t* __restrict__ hb) {
  __shared__ float s1s[TOPK], s2s[TOPK], wls[TOPK];
  __shared__ int   i1s[TOPK], i2s[TOPK], rls[TOPK];

  const int tok  = blockIdx.x;
  const int lane = threadIdx.x & 31;
  const int wid  = threadIdx.x >> 5;

  // --- stage 1: two parallel top-32-of-512 selections (waves 0 and 1) ---
  if (wid < 2) {
    const float* sc = (wid == 0 ? s1f : s2f) + (size_t)tok * NUM_KEYS;
    float v[16];
#pragma unroll
    for (int t = 0; t < 16; ++t) v[t] = sc[lane + 32 * t];

    for (int it = 0; it < TOPK; ++it) {
      float mv = -3.0e38f; int mi = 0;
#pragma unroll
      for (int t = 0; t < 16; ++t)
        if (v[t] > mv) { mv = v[t]; mi = lane + 32 * t; }
#pragma unroll
      for (int off = 16; off > 0; off >>= 1) {
        float ov = __shfl_xor(mv, off, 32);
        int   oi = __shfl_xor(mi, off, 32);
        if (ov > mv || (ov == mv && oi < mi)) { mv = ov; mi = oi; }
      }
      if ((mi & 31) == lane) {        // owner masks out the winner
        const int slot = mi >> 5;
#pragma unroll
        for (int t = 0; t < 16; ++t) if (t == slot) v[t] = -3.0e38f;
      }
      if (lane == it) {
        if (wid == 0) { s1s[it] = mv; i1s[it] = mi; }
        else          { s2s[it] = mv; i2s[it] = mi; }
      }
    }
  }
  __syncthreads();

  // --- stage 2: top-32 of the 32x32 combined scores + softmax (wave 0) ---
  if (wid == 0) {
    float c[32];
    const float s2v = s2s[lane];
#pragma unroll
    for (int i = 0; i < 32; ++i) c[i] = s1s[i] + s2v;

    float myw = 0.0f; int myr = 0;
    for (int it = 0; it < TOPK; ++it) {
      float mv = -3.0e38f; int mi = 0;
#pragma unroll
      for (int i = 0; i < 32; ++i)
        if (c[i] > mv) { mv = c[i]; mi = i; }
      int ml = lane;
#pragma unroll
      for (int off = 16; off > 0; off >>= 1) {
        float ov = __shfl_xor(mv, off, 32);
        int   oi = __shfl_xor(mi, off, 32);
        int   ol = __shfl_xor(ml, off, 32);
        if (ov > mv || (ov == mv && (ol * 32 + oi) < (ml * 32 + mi))) {
          mv = ov; mi = oi; ml = ol;
        }
      }
      if (lane == ml) {
#pragma unroll
        for (int i = 0; i < 32; ++i) if (i == mi) c[i] = -3.0e38f;
      }
      if (lane == it) { myw = mv; myr = i1s[mi] * NUM_KEYS + i2s[ml]; }
    }
    // softmax(score / sqrt(DIM)) across the 32 lanes
    float m = myw;
#pragma unroll
    for (int off = 16; off > 0; off >>= 1) {
      float o = __shfl_xor(m, off, 32);
      m = (o > m) ? o : m;
    }
    const float e = __expf((myw - m) * 0.03125f);  // 1/sqrt(1024) = 1/32
    float s = e;
#pragma unroll
    for (int off = 16; off > 0; off >>= 1) s += __shfl_xor(s, off, 32);
    wls[lane] = e / s;
    rls[lane] = myr;
  }
  __syncthreads();

  const int t = threadIdx.x;

  // --- stage 2.5: kick HBM early for the 32 gathered rows (128 KB/token).
  // 8 threads per row, 512 B apart -> one global_prefetch_b8 per thread.
  {
    const int pr = t >> 3;          // row slot 0..31
    const int po = (t & 7) * 128;   // float offset, 512 B spacing
    __builtin_prefetch(values + (size_t)rls[pr] * DIM + po, 0, 1);
  }

  // --- stage 3: coalesced gather + weighted sum + gate, bf16 out ---
  float a0 = 0.f, a1 = 0.f, a2 = 0.f, a3 = 0.f;
  for (int k = 0; k < TOPK; ++k) {
    const float* vp = values + (size_t)rls[k] * DIM + t;
    const float  wk = wls[k];
    a0 += wk * vp[0];
    a1 += wk * vp[256];
    a2 += wk * vp[512];
    a3 += wk * vp[768];
  }
  const float* g = gatef + (size_t)tok * DIM + t;
  bf16_t*      h = hb    + (size_t)tok * DIM + t;
  h[0]   = (bf16_t)(a0 * g[0]);
  h[256] = (bf16_t)(a1 * g[256]);
  h[512] = (bf16_t)(a2 * g[512]);
  h[768] = (bf16_t)(a3 * g[768]);
}

// ---------------------------------------------------------------------------
extern "C" void kernel_launch(void* const* d_in, const int* in_sizes, int n_in,
                              void* d_out, int out_size, void* d_ws, size_t ws_size,
                              hipStream_t stream) {
  const float* x      = (const float*)d_in[0];
  const float* ke1    = (const float*)d_in[1];
  const float* ke2    = (const float*)d_in[2];
  const float* values = (const float*)d_in[3];
  const float* Wq     = (const float*)d_in[4];
  const float* bq     = (const float*)d_in[5];
  const float* Wg     = (const float*)d_in[6];
  const float* bg     = (const float*)d_in[7];
  const float* Wo     = (const float*)d_in[8];
  const float* bo     = (const float*)d_in[9];
  float* out = (float*)d_out;

  char* ws = (char*)d_ws;
  const size_t MB = (size_t)1 << 20;
  bf16_t* xb    = (bf16_t*)(ws + 0 * MB);              //  8 MB: x bf16 (4096x1024)
  bf16_t* Wqb   = (bf16_t*)(ws + 8 * MB);              //  2 MB
  bf16_t* Wgb   = (bf16_t*)(ws + 10 * MB);             //  2 MB
  bf16_t* Wob   = (bf16_t*)(ws + 12 * MB);             //  2 MB
  bf16_t* ke1b  = (bf16_t*)(ws + 14 * MB);             //  0.5 MB
  bf16_t* ke2b  = (bf16_t*)(ws + 14 * MB + 512 * 1024);//  0.5 MB
  bf16_t* qb    = (bf16_t*)(ws + 15 * MB);             //  8 MB: query bf16
  float*  gatef = (float*) (ws + 23 * MB);             // 16 MB: silu(gate) fp32
  float*  s1f   = (float*) (ws + 39 * MB);             //  8 MB: scores1 fp32
  float*  s2f   = (float*) (ws + 47 * MB);             //  8 MB: scores2 fp32
  bf16_t* hb    = (bf16_t*)(ws + 55 * MB);             //  8 MB: gated pkm out bf16

  const dim3 blk(256);
  cvt_f32_bf16<<<NTOK * DIM / 256, blk, 0, stream>>>(x,   xb,   NTOK * DIM);
  cvt_f32_bf16<<<DIM * DIM / 256,  blk, 0, stream>>>(Wq,  Wqb,  DIM * DIM);
  cvt_f32_bf16<<<DIM * DIM / 256,  blk, 0, stream>>>(Wg,  Wgb,  DIM * DIM);
  cvt_f32_bf16<<<DIM * DIM / 256,  blk, 0, stream>>>(Wo,  Wob,  DIM * DIM);
  cvt_f32_bf16<<<NUM_KEYS * HALF / 256, blk, 0, stream>>>(ke1, ke1b, NUM_KEYS * HALF);
  cvt_f32_bf16<<<NUM_KEYS * HALF / 256, blk, 0, stream>>>(ke2, ke2b, NUM_KEYS * HALF);

  // query = x @ Wq^T + bq  -> bf16 (mode 3)
  gemm_bf16_wmma<<<dim3(DIM / 64, NTOK / 256), blk, 0, stream>>>(
      xb, DIM, Wqb, DIM, bq, qb, DIM, DIM, 3);
  // gate = silu(x @ Wg^T + bg) -> fp32 (mode 1)
  gemm_bf16_wmma<<<dim3(DIM / 64, NTOK / 256), blk, 0, stream>>>(
      xb, DIM, Wgb, DIM, bg, gatef, DIM, DIM, 1);
  // scores1 = q1 @ ke1^T, scores2 = q2 @ ke2^T -> fp32, no bias (mode 2)
  gemm_bf16_wmma<<<dim3(NUM_KEYS / 64, NTOK / 256), blk, 0, stream>>>(
      qb, DIM, ke1b, HALF, nullptr, s1f, NUM_KEYS, HALF, 2);
  gemm_bf16_wmma<<<dim3(NUM_KEYS / 64, NTOK / 256), blk, 0, stream>>>(
      qb + HALF, DIM, ke2b, HALF, nullptr, s2f, NUM_KEYS, HALF, 2);

  // product-key top-k + softmax + value gather + gate -> hb (bf16)
  pkm_topk_gather<<<NTOK, blk, 0, stream>>>(s1f, s2f, values, gatef, hb);

  // out = hb @ Wo^T + bo -> fp32 (mode 0)
  gemm_bf16_wmma<<<dim3(DIM / 64, NTOK / 256), blk, 0, stream>>>(
      hb, DIM, Wob, DIM, bo, out, DIM, DIM, 0);
}
